// YOLO_24343874633765
// MI455X (gfx1250) — matrix operand order repaired and checked by reading, not compiled, not run on hardware
//
#include <hip/hip_runtime.h>
#include <hip/hip_bf16.h>

typedef __attribute__((ext_vector_type(16))) __bf16 v16bf;
typedef __attribute__((ext_vector_type(8)))  float  v8f;

#define LEAKY(v) ((v) > 0.f ? (v) : 0.1f * (v))

static __device__ __forceinline__ unsigned short bf16_bits(float f) {
    union { __bf16 b; unsigned short u; } c;
    c.b = (__bf16)f;
    return c.u;
}

// ---------------------------------------------------------------------------
// BN folding: y = scale[c]*conv + shift[c]
// ---------------------------------------------------------------------------
__global__ void bn_prep_kernel(const float* __restrict__ g, const float* __restrict__ b,
                               const float* __restrict__ m, const float* __restrict__ v,
                               float* __restrict__ scale, float* __restrict__ shift, int C) {
    int i = blockIdx.x * blockDim.x + threadIdx.x;
    if (i < C) {
        float s = g[i] * rsqrtf(v[i] + 1e-5f);
        scale[i] = s;
        shift[i] = b[i] - m[i] * s;
    }
}

__global__ void bias_prep_kernel(const float* __restrict__ b,
                                 float* __restrict__ scale, float* __restrict__ shift, int C) {
    int i = blockIdx.x * blockDim.x + threadIdx.x;
    if (i < C) { scale[i] = 1.f; shift[i] = b[i]; }
}

// ---------------------------------------------------------------------------
// Implicit-GEMM conv + folded BN/bias + optional LeakyReLU.
// Block = 128 threads (4 waves); tile = 64(Cout) x 32(spatial).
// Each wave holds TWO 16x16 accumulators (two spatial sub-tiles) so the A
// (weight) fragment is loaded once per chunk but feeds two WMMAs.
// B tile (32K x 32N, shared by all 4 waves) staged in LDS as [N][K] bf16,
// double-buffered; one split-barrier per K-chunk; ds_load_b128 fragment reads;
// one ds_store_b128 per thread for staging. All global loads unconditional
// with clamped addresses (validity via v_cndmask, no exec divergence).
// Fragment packing per CDNA5 ISA 7.12.2 (wave32):
//   A: lane L -> M = L&15 ; K = 8*(L>>4) + 16*(j>>3) + (j&7)
//   B: lane L -> N = L&15 ; K = 8*(L>>4) + 16*(j>>3) + (j&7)
//   C: VGPR r -> M = r + 8*(L>>4) ; N = L&15
// ---------------------------------------------------------------------------
template <int KS, bool TAIL>
__global__ void conv_wmma_kernel(const float* __restrict__ in, const float* __restrict__ wgt,
                                 const float* __restrict__ scale, const float* __restrict__ shift,
                                 float* __restrict__ out,
                                 int Cin, int Cout, int H, int W,
                                 int Hout, int Wout, int stride, int pad, int leaky) {
    __shared__ __align__(16) unsigned short Bsh[2][1024];  // [buf][N=32][K=32]

    const int tid  = threadIdx.x;
    const int lane = tid & 31;
    const int wv   = tid >> 5;                 // wave id 0..3
    const int n    = blockIdx.z;
    const int co0  = blockIdx.y * 64 + wv * 16;
    const int sp0  = blockIdx.x * 32;
    const int HW   = Hout * Wout;
    const int Ktot = Cin * KS * KS;

    const int mrow = lane & 15;
    const int half = lane >> 4;
    const int colf = lane & 15;                // fragment column within sub-tile
    const int spf0 = sp0 + colf;               // sub-tile 0 spatial
    const int spf1 = sp0 + colf + 16;          // sub-tile 1 spatial

    // clamped weight row (OOB rows computed but never stored)
    const int co_c = min(co0 + mrow, Cout - 1);
    const float* wrow = wgt + (long)co_c * Ktot;

    // staging coordinates: thread handles 8 consecutive K of one column
    const int scol = tid >> 2;                 // 0..31
    const int sk0  = (tid & 3) * 8;            // 0,8,16,24
    const int ssp  = sp0 + scol;
    const int sspc = min(ssp, HW - 1);
    const int soy  = sspc / Wout;
    const int sox  = sspc - soy * Wout;
    const long inb = (long)n * Cin * H * W;

    v8f acc0 = {};
    v8f acc1 = {};
    int p = 0;
    for (int kc = 0; kc < Ktot; kc += 32, p ^= 1) {
        // ---- stage B tile into LDS (cooperative, branchless) ----
        unsigned short hb[8];
        #pragma unroll
        for (int i = 0; i < 8; ++i) {
            int kk  = kc + sk0 + i;
            int kkc = TAIL ? min(kk, Ktot - 1) : kk;
            int ci, ky, kx;
            if (KS == 1) { ci = kkc; ky = 0; kx = 0; }
            else         { ci = kkc / 9; int r = kkc - ci * 9; ky = r / 3; kx = r - ky * 3; }
            int iy = soy * stride - pad + ky;
            int ix = sox * stride - pad + kx;
            bool valid = (iy >= 0) & (iy < H) & (ix >= 0) & (ix < W) & (ssp < HW);
            if (TAIL) valid = valid & (kk < Ktot);
            int iyc = min(max(iy, 0), H - 1);
            int ixc = min(max(ix, 0), W - 1);
            float x = in[inb + ((long)ci * H + iyc) * W + ixc];  // unconditional
            hb[i] = bf16_bits(valid ? x : 0.f);
        }
        uint4 pk;
        pk.x = (unsigned int)hb[0] | ((unsigned int)hb[1] << 16);
        pk.y = (unsigned int)hb[2] | ((unsigned int)hb[3] << 16);
        pk.z = (unsigned int)hb[4] | ((unsigned int)hb[5] << 16);
        pk.w = (unsigned int)hb[6] | ((unsigned int)hb[7] << 16);
        ((uint4*)Bsh[p])[tid] = pk;                       // ds_store_b128

        // ---- A fragment: weights (contiguous in K) ----
        v16bf a;
        if (TAIL) {
            #pragma unroll
            for (int j = 0; j < 16; ++j) {
                int kk  = kc + half * 8 + ((j >> 3) << 4) + (j & 7);
                int kkc = min(kk, Ktot - 1);
                float wvv = wrow[kkc];                    // unconditional
                a[j] = (__bf16)(kk < Ktot ? wvv : 0.f);
            }
        } else {
            if (kc + 32 < Ktot) __builtin_prefetch(wrow + kc + 32, 0, 0);
            const float4* p0 = (const float4*)(wrow + kc + half * 8);
            const float4* p1 = (const float4*)(wrow + kc + half * 8 + 16);
            float4 w0 = p0[0], w1 = p0[1];                // K = 0..7
            float4 w2 = p1[0], w3 = p1[1];                // K = 16..23
            float af[16] = { w0.x, w0.y, w0.z, w0.w, w1.x, w1.y, w1.z, w1.w,
                             w2.x, w2.y, w2.z, w2.w, w3.x, w3.y, w3.z, w3.w };
            #pragma unroll
            for (int j = 0; j < 16; ++j) a[j] = (__bf16)af[j];
        }

        __syncthreads();                                  // B tile ready

        // ---- two B fragments from LDS: 4 x ds_load_b128 ----
        union { v16bf v; uint4 q[2]; } b0, b1;
        const uint4* bp0 = (const uint4*)(Bsh[p] + colf * 32 + half * 8);
        const uint4* bp1 = (const uint4*)(Bsh[p] + (colf + 16) * 32 + half * 8);
        b0.q[0] = bp0[0];
        b0.q[1] = bp0[2];
        b1.q[0] = bp1[0];
        b1.q[1] = bp1[2];

        acc0 = __builtin_amdgcn_wmma_f32_16x16x32_bf16(false, a, false, b0.v,
                                                       (short)0, acc0, false, false);
        acc1 = __builtin_amdgcn_wmma_f32_16x16x32_bf16(false, a, false, b1.v,
                                                       (short)0, acc1, false, false);
    }

    #pragma unroll
    for (int r = 0; r < 8; ++r) {
        int oc = co0 + r + half * 8;
        if (oc < Cout) {
            float s = scale[oc], sh = shift[oc];
            long base = ((long)n * Cout + oc) * HW;
            if (spf0 < HW) {
                float y = s * acc0[r] + sh;
                if (leaky) y = LEAKY(y);
                out[base + spf0] = y;
            }
            if (spf1 < HW) {
                float y = s * acc1[r] + sh;
                if (leaky) y = LEAKY(y);
                out[base + spf1] = y;
            }
        }
    }
}

// ---------------------------------------------------------------------------
// FC: out[b][o] = act( in[b][:] . wgt[o][:] + bias[o] ).
// Block = 128 threads (4 waves); tile = 64 outputs x 16 (batch padded).
// Weight-streaming bound; Ktot multiple of 32 for all FC layers here.
// ---------------------------------------------------------------------------
__global__ void fc_wmma_kernel(const float* __restrict__ in, const float* __restrict__ wgt,
                               const float* __restrict__ bias, float* __restrict__ out,
                               int Nb, int O, int Ktot, int leaky) {
    __shared__ __align__(16) unsigned short Bsh[2][512];

    const int tid  = threadIdx.x;
    const int lane = tid & 31;
    const int wv   = tid >> 5;
    const int o0   = blockIdx.x * 64 + wv * 16;
    const int mrow = lane & 15;
    const int half = lane >> 4;
    const int colf = lane & 15;                 // batch index

    const int oc_c = min(o0 + mrow, O - 1);
    const float* wrow = wgt + (long)oc_c * Ktot;

    const int scol = tid >> 3;
    const int sk0  = (tid & 7) * 4;
    const int scolc = min(scol, Nb - 1);

    v8f acc = {};
    int p = 0;
    for (int kc = 0; kc < Ktot; kc += 32, p ^= 1) {
        unsigned short hb[4];
        #pragma unroll
        for (int i = 0; i < 4; ++i) {
            float x = in[(long)scolc * Ktot + kc + sk0 + i];    // unconditional
            hb[i] = bf16_bits(scol < Nb ? x : 0.f);
        }
        uint2 pk;
        pk.x = (unsigned int)hb[0] | ((unsigned int)hb[1] << 16);
        pk.y = (unsigned int)hb[2] | ((unsigned int)hb[3] << 16);
        ((uint2*)Bsh[p])[tid] = pk;

        if (kc + 32 < Ktot) __builtin_prefetch(wrow + kc + 32, 0, 0);
        const float4* p0 = (const float4*)(wrow + kc + half * 8);
        const float4* p1 = (const float4*)(wrow + kc + half * 8 + 16);
        float4 w0 = p0[0], w1 = p0[1];
        float4 w2 = p1[0], w3 = p1[1];
        float af[16] = { w0.x, w0.y, w0.z, w0.w, w1.x, w1.y, w1.z, w1.w,
                         w2.x, w2.y, w2.z, w2.w, w3.x, w3.y, w3.z, w3.w };
        v16bf a;
        #pragma unroll
        for (int j = 0; j < 16; ++j) a[j] = (__bf16)af[j];

        __syncthreads();

        union { v16bf v; uint4 q[2]; } bu;
        const uint4* bp = (const uint4*)(Bsh[p] + colf * 32 + half * 8);
        bu.q[0] = bp[0];
        bu.q[1] = bp[2];

        acc = __builtin_amdgcn_wmma_f32_16x16x32_bf16(false, a, false, bu.v,
                                                      (short)0, acc, false, false);
    }

    #pragma unroll
    for (int r = 0; r < 8; ++r) {
        int oc = o0 + r + half * 8;
        if (oc < O && colf < Nb) {
            float y = acc[r] + bias[oc];
            if (leaky) y = LEAKY(y);
            out[(long)colf * O + oc] = y;
        }
    }
}

// ---------------------------------------------------------------------------
// 2x2 max pool, stride 2 (N fixed = 4)
// ---------------------------------------------------------------------------
__global__ void maxpool_kernel(const float* __restrict__ in, float* __restrict__ out,
                               int C, int H, int W) {
    int Ho = H / 2, Wo = W / 2;
    long total = 4L * C * Ho * Wo;
    long i = (long)blockIdx.x * blockDim.x + threadIdx.x;
    if (i >= total) return;
    int x = (int)(i % Wo); long t = i / Wo;
    int y = (int)(t % Ho); t /= Ho;
    int c = (int)(t % C);  int n = (int)(t / C);
    const float* p = in + (((long)n * C + c) * H + 2 * y) * W + 2 * x;
    out[i] = fmaxf(fmaxf(p[0], p[1]), fmaxf(p[W], p[W + 1]));
}

// ---------------------------------------------------------------------------
// Decode: softmax cls, sigmoid obj/offsets, box corners (clipped to [0,448])
// ---------------------------------------------------------------------------
__global__ void decode_kernel(const float* __restrict__ cls, const float* __restrict__ resp,
                              const float* __restrict__ off,
                              float* __restrict__ boxes, float* __restrict__ scores,
                              float* __restrict__ labels) {
    const int S = 7, BBn = 2, NC = 20;
    const float IMG = 448.f;
    int idx = blockIdx.x * blockDim.x + threadIdx.x;
    int total = 4 * S * S * BBn;
    if (idx >= total) return;
    int bb   = idx % BBn;
    int cell = (idx / BBn) % (S * S);
    int bi   = idx / (BBn * S * S);
    int gy = cell / S, gx = cell % S;

    const float* cl = cls + (long)bi * NC * S * S + cell;
    float mx = -1e30f; int lab = 0;
    for (int c = 0; c < NC; ++c) { float v = cl[c * S * S]; if (v > mx) { mx = v; lab = c; } }
    float sum = 0.f, pm = 0.f;
    for (int c = 0; c < NC; ++c) { float e = expf(cl[c * S * S] - mx); sum += e; if (c == lab) pm = e; }
    float clsmax = pm / sum;

    float obj = 1.f / (1.f + expf(-resp[((long)bi * BBn + bb) * S * S + cell]));

    const float* t = off + ((long)bi * BBn * 4 + bb * 4) * S * S + cell;
    float t0 = t[0], t1 = t[S * S], t2 = t[2 * S * S], t3 = t[3 * S * S];
    float cx = (1.f / (1.f + expf(-t0)) + (float)gx) / S * IMG;
    float cy = (1.f / (1.f + expf(-t1)) + (float)gy) / S * IMG;
    float w  = t2 * t2 * IMG;
    float h  = t3 * t3 * IMG;
    float x1 = fminf(fmaxf(cx - 0.5f * w, 0.f), IMG);
    float y1 = fminf(fmaxf(cy - 0.5f * h, 0.f), IMG);
    float x2 = fminf(fmaxf(cx + 0.5f * w, 0.f), IMG);
    float y2 = fminf(fmaxf(cy + 0.5f * h, 0.f), IMG);

    int nn = cell * BBn + bb;
    float* bx = boxes + ((long)bi * 98 + nn) * 4;
    bx[0] = x1; bx[1] = y1; bx[2] = x2; bx[3] = y2;
    scores[bi * 98 + nn] = obj * clsmax;
    labels[bi * 98 + nn] = (float)lab;
}

// ---------------------------------------------------------------------------
// NMS: one block per image; LDS rank-sort then sequential suppression.
// Output layout: boxes[4][98][4] | scores[4][98] | labels[4][98]
// ---------------------------------------------------------------------------
__global__ void nms_kernel(const float* __restrict__ boxes, const float* __restrict__ scores,
                           const float* __restrict__ labels, float* __restrict__ out) {
    const int N = 98;
    const float CONF = 0.02f, THR = 0.5f;
    int b = blockIdx.x, tid = threadIdx.x;

    __shared__ float usc[N];
    __shared__ float sx1[N], sy1[N], sx2[N], sy2[N], ssc[N], slb[N], sar[N];
    __shared__ int   skeep[N];

    if (tid < N) usc[tid] = scores[b * N + tid];
    __syncthreads();

    if (tid < N) {
        float s = usc[tid];
        int rank = 0;
        for (int j = 0; j < N; ++j) {
            float sj = usc[j];
            rank += (sj > s) || (sj == s && j < tid);
        }
        const float* bx = boxes + ((long)b * N + tid) * 4;
        sx1[rank] = bx[0]; sy1[rank] = bx[1]; sx2[rank] = bx[2]; sy2[rank] = bx[3];
        ssc[rank] = s;     slb[rank] = labels[b * N + tid];
    }
    __syncthreads();

    if (tid < N) {
        sar[tid]   = (sx2[tid] - sx1[tid]) * (sy2[tid] - sy1[tid]);
        skeep[tid] = (ssc[tid] > CONF) ? 1 : 0;
    }

    for (int i = 0; i < N; ++i) {
        __syncthreads();
        int ki = skeep[i];
        if (tid < N && tid > i && ki) {
            float ix1 = fmaxf(sx1[i], sx1[tid]);
            float iy1 = fmaxf(sy1[i], sy1[tid]);
            float ix2 = fminf(sx2[i], sx2[tid]);
            float iy2 = fminf(sy2[i], sy2[tid]);
            float inter = fmaxf(ix2 - ix1, 0.f) * fmaxf(iy2 - iy1, 0.f);
            float iou   = inter / (sar[i] + sar[tid] - inter + 1e-9f);
            if (iou > THR) skeep[tid] = 0;
        }
    }
    __syncthreads();

    if (tid < N) {
        int k = skeep[tid];
        float* ob = out + ((long)b * N + tid) * 4;
        ob[0] = k ? sx1[tid] : 0.f;
        ob[1] = k ? sy1[tid] : 0.f;
        ob[2] = k ? sx2[tid] : 0.f;
        ob[3] = k ? sy2[tid] : 0.f;
        out[4 * N * 4 + b * N + tid]         = k ? ssc[tid] : 0.f;
        out[4 * N * 4 + 4 * N + b * N + tid] = k ? slb[tid] : -1.f;
    }
}

// ---------------------------------------------------------------------------
// Host orchestration
// ---------------------------------------------------------------------------
static void launch_conv(hipStream_t s, const float* in, float* out, const float* w,
                        const float* scale, const float* shift,
                        int Cin, int Cout, int H, int stride, int ks, int leaky) {
    int Hout = (stride == 2) ? H / 2 : H;            // SAME, H even
    int pad  = (ks == 3 && stride == 1) ? 1 : 0;     // SAME pad_lo (stride2: lo=0)
    dim3 grid((unsigned)((Hout * Hout + 31) / 32), (unsigned)((Cout + 63) / 64), 4);
    bool tail = ((Cin * ks * ks) & 31) != 0;
    if (ks == 3) {
        if (tail)
            conv_wmma_kernel<3, true><<<grid, 128, 0, s>>>(in, w, scale, shift, out,
                Cin, Cout, H, H, Hout, Hout, stride, pad, leaky);
        else
            conv_wmma_kernel<3, false><<<grid, 128, 0, s>>>(in, w, scale, shift, out,
                Cin, Cout, H, H, Hout, Hout, stride, pad, leaky);
    } else {
        conv_wmma_kernel<1, false><<<grid, 128, 0, s>>>(in, w, scale, shift, out,
            Cin, Cout, H, H, Hout, Hout, stride, pad, leaky);
    }
}

extern "C" void kernel_launch(void* const* d_in, const int* in_sizes, int n_in,
                              void* d_out, int out_size, void* d_ws, size_t ws_size,
                              hipStream_t stream) {
    (void)in_sizes; (void)n_in; (void)out_size; (void)ws_size;

    const float* x = (const float*)d_in[0];
    float* ws = (float*)d_ws;

    const size_t BUF = 25700000;     // >= 4*32*448*448 = 25,690,112
    float* bufA = ws;
    float* bufB = ws + BUF;
    size_t o = 2 * BUF;
    float* scl  = ws + o; o += 1024;
    float* shf  = ws + o; o += 1024;
    float* fco  = ws + o; o += 4 * 4096;
    float* clsb = ws + o; o += 4 * 980;
    float* rspb = ws + o; o += 4 * 98;
    float* offb = ws + o; o += 4 * 392;
    float* bxb  = ws + o; o += 4 * 98 * 4;
    float* scb  = ws + o; o += 4 * 98;
    float* lbb  = ws + o; o += 4 * 98;

    const int seq[][2] = {
        {32,3},{-1,0},{64,3},{-1,0},{128,3},{64,1},{128,3},{-1,0},
        {256,3},{128,1},{256,3},{-1,0},{512,3},{256,1},{512,3},{256,1},{512,3},{-1,0},
        {1024,3},{512,1},{1024,3},{512,1},{1024,3}
    };

    const float* cur = x;
    float* dst = bufA;
    float* alt = bufB;
    int H = 448, Cin = 3, pi = 1;

    for (int s = 0; s < (int)(sizeof(seq) / sizeof(seq[0])); ++s) {
        if (seq[s][0] == -1) {
            long total = 4L * Cin * (H / 2) * (H / 2);
            maxpool_kernel<<<(unsigned)((total + 255) / 256), 256, 0, stream>>>(cur, dst, Cin, H, H);
            H /= 2;
        } else {
            int cout = seq[s][0], ks = seq[s][1];
            const float* w = (const float*)d_in[pi + 0];
            bn_prep_kernel<<<(cout + 255) / 256, 256, 0, stream>>>(
                (const float*)d_in[pi + 1], (const float*)d_in[pi + 2],
                (const float*)d_in[pi + 3], (const float*)d_in[pi + 4], scl, shf, cout);
            pi += 5;
            launch_conv(stream, cur, dst, w, scl, shf, Cin, cout, H, 1, ks, 1);
            Cin = cout;
        }
        cur = dst; float* t = dst; dst = alt; alt = t;
    }

    // local conv layers (bias, leaky; first has stride 2): H 14 -> 7
    for (int i = 0; i < 4; ++i) {
        const float* w = (const float*)d_in[pi + 0];
        const float* b = (const float*)d_in[pi + 1];
        pi += 2;
        bias_prep_kernel<<<(1024 + 255) / 256, 256, 0, stream>>>(b, scl, shf, 1024);
        int stride = (i == 0) ? 2 : 1;
        launch_conv(stream, cur, dst, w, scl, shf, 1024, 1024, H, stride, 3, 1);
        if (stride == 2) H /= 2;
        cur = dst; float* t = dst; dst = alt; alt = t;
    }

    // cur = [4,1024,7,7] == [4,50176] row-major
    const int KFC = 1024 * 7 * 7;
    fc_wmma_kernel<<<(4096 + 63) / 64, 128, 0, stream>>>(
        cur, (const float*)d_in[pi + 0], (const float*)d_in[pi + 1], fco, 4, 4096, KFC, 1);
    pi += 2;
    fc_wmma_kernel<<<(980 + 63) / 64, 128, 0, stream>>>(
        fco, (const float*)d_in[pi + 0], (const float*)d_in[pi + 1], clsb, 4, 980, 4096, 0);
    pi += 2;
    fc_wmma_kernel<<<(98 + 63) / 64, 128, 0, stream>>>(
        fco, (const float*)d_in[pi + 0], (const float*)d_in[pi + 1], rspb, 4, 98, 4096, 0);
    pi += 2;
    fc_wmma_kernel<<<(392 + 63) / 64, 128, 0, stream>>>(
        fco, (const float*)d_in[pi + 0], (const float*)d_in[pi + 1], offb, 4, 392, 4096, 0);
    pi += 2;

    decode_kernel<<<(4 * 49 * 2 + 127) / 128, 128, 0, stream>>>(clsb, rspb, offb, bxb, scb, lbb);
    nms_kernel<<<4, 128, 0, stream>>>(bxb, scb, lbb, (float*)d_out);
}
